// DensitySetAbstraction_45612552683640
// MI455X (gfx1250) — matrix operand order repaired and checked by reading, not compiled, not run on hardware
//
#include <hip/hip_runtime.h>
#include <hip/hip_bf16.h>
#include <cstdint>
#include <cstddef>

#define BB 4
#define NN 4096
#define DD 64
#define NPOINTS 1024
#define NSAMP 32
#define KS (NSAMP * NPOINTS)   // 32768 flattened (k,s) columns
#define CIN0 (3 + DD)          // 67
#define CIN0P 96               // padded to multiple of 32
#define EPSV 1e-5f

typedef __attribute__((ext_vector_type(16))) _Float16 v16h;
typedef __attribute__((ext_vector_type(8)))  _Float16 v8h;
typedef __attribute__((ext_vector_type(8)))  float    v8f;

// ---------------------------------------------------------------------------
// Zero fill (for channel padding regions)
// ---------------------------------------------------------------------------
__global__ void zero_kernel(float* __restrict__ p, size_t n) {
  size_t i = (size_t)blockIdx.x * 256 + threadIdx.x;
  if (i < n) p[i] = 0.0f;
}

// ---------------------------------------------------------------------------
// Stage weights: W (O x C f32) -> Wh (Opad x Cpad f16), zero padded
// ---------------------------------------------------------------------------
__global__ void prep_weights_kernel(const float* __restrict__ W, _Float16* __restrict__ Wh,
                                    int O, int C, int Opad, int Cpad) {
  int t = blockIdx.x * 256 + threadIdx.x;
  if (t >= Opad * Cpad) return;
  int o = t / Cpad, c = t % Cpad;
  Wh[t] = (o < O && c < C) ? (_Float16)W[(size_t)o * C + c] : (_Float16)0.0f;
}

// ---------------------------------------------------------------------------
// Stage activations: (B, Cp, Npb) f32 -> (B, Npb, Cp) f16 (tiled transpose)
// grid (Npb/32, Cp/32, B), block (32, 8)
// ---------------------------------------------------------------------------
__global__ void __launch_bounds__(256) act_transpose_kernel(
    const float* __restrict__ in, _Float16* __restrict__ out, int Cp, int Npb) {
  __shared__ float tile[32][33];
  const int b  = blockIdx.z;
  const int n0 = blockIdx.x * 32;
  const int c0 = blockIdx.y * 32;
  const float* ib = in + (size_t)b * Cp * Npb;
  _Float16* ob = out + (size_t)b * Npb * Cp;
  #pragma unroll
  for (int cc = threadIdx.y; cc < 32; cc += 8)
    tile[cc][threadIdx.x] = ib[(size_t)(c0 + cc) * Npb + n0 + threadIdx.x];
  __syncthreads();
  #pragma unroll
  for (int nn = threadIdx.y; nn < 32; nn += 8)
    ob[(size_t)(n0 + nn) * Cp + c0 + threadIdx.x] = (_Float16)tile[threadIdx.x][nn];
}

// ---------------------------------------------------------------------------
// Gaussian kernel density
// ---------------------------------------------------------------------------
__global__ void density_kernel(const float* __restrict__ xyz, float* __restrict__ dens) {
  const int b = blockIdx.y;
  const int n = blockIdx.x * 256 + threadIdx.x;
  const float* xb = xyz + (size_t)b * 3 * NN;
  __shared__ float tx[256], ty[256], tz[256];
  const float px = xb[n], py = xb[NN + n], pz = xb[2 * NN + n];
  float acc = 0.0f;
  for (int m0 = 0; m0 < NN; m0 += 256) {
    tx[threadIdx.x] = xb[m0 + threadIdx.x];
    ty[threadIdx.x] = xb[NN + m0 + threadIdx.x];
    tz[threadIdx.x] = xb[2 * NN + m0 + threadIdx.x];
    __syncthreads();
    #pragma unroll 4
    for (int j = 0; j < 256; ++j) {
      float dx = px - tx[j], dy = py - ty[j], dz = pz - tz[j];
      acc += __expf(-50.0f * (dx * dx + dy * dy + dz * dz));
    }
    __syncthreads();
  }
  dens[(size_t)b * 32 * NN + n] = acc * (1.0f / (float)NN) * 4.0f;  // ch0 of padded buf
}

// ---------------------------------------------------------------------------
// Farthest point sampling: one 1024-thread block per batch
// ---------------------------------------------------------------------------
__global__ void __launch_bounds__(1024) fps_kernel(const float* __restrict__ xyz,
                                                   int* __restrict__ fpsIdx) {
  const int b = blockIdx.x;
  const int tid = threadIdx.x;
  __shared__ float sx[NN], sy[NN], sz[NN];
  __shared__ float rv[32];
  __shared__ int   ri[32];
  __shared__ int   s_far;
  const float* xb = xyz + (size_t)b * 3 * NN;
  for (int i = tid; i < NN; i += 1024) {
    sx[i] = xb[i]; sy[i] = xb[NN + i]; sz[i] = xb[2 * NN + i];
  }
  float dreg[4];
  #pragma unroll
  for (int j = 0; j < 4; ++j) dreg[j] = 1.0e10f;
  if (tid == 0) s_far = 0;
  __syncthreads();
  const int wv = tid >> 5, ln = tid & 31;
  for (int it = 0; it < NPOINTS; ++it) {
    const int far = s_far;
    if (tid == 0) fpsIdx[(size_t)b * NPOINTS + it] = far;
    const float fx = sx[far], fy = sy[far], fz = sz[far];
    float bv = -1.0e30f; int bi = 0;
    #pragma unroll
    for (int j = 0; j < 4; ++j) {
      const int i = tid + j * 1024;
      float dx = sx[i] - fx, dy = sy[i] - fy, dz = sz[i] - fz;
      float dd = fminf(dreg[j], dx * dx + dy * dy + dz * dz);
      dreg[j] = dd;
      if (dd > bv) { bv = dd; bi = i; }
    }
    #pragma unroll
    for (int off = 16; off > 0; off >>= 1) {
      float ov = __shfl_xor(bv, off, 32);
      int   oi = __shfl_xor(bi, off, 32);
      if (ov > bv || (ov == bv && oi < bi)) { bv = ov; bi = oi; }
    }
    if (ln == 0) { rv[wv] = bv; ri[wv] = bi; }
    __syncthreads();
    if (wv == 0) {
      float v2 = rv[ln]; int i2 = ri[ln];
      #pragma unroll
      for (int off = 16; off > 0; off >>= 1) {
        float ov = __shfl_xor(v2, off, 32);
        int   oi = __shfl_xor(i2, off, 32);
        if (ov > v2 || (ov == v2 && oi < i2)) { v2 = ov; i2 = oi; }
      }
      if (ln == 0) s_far = i2;
    }
    __syncthreads();
  }
}

// ---------------------------------------------------------------------------
// Gather FPS results
// ---------------------------------------------------------------------------
__global__ void gather_fps_kernel(const float* __restrict__ xyz, const int* __restrict__ labels,
                                  const int* __restrict__ fpsIdx, float* __restrict__ newxyz,
                                  float* __restrict__ out_xyz, float* __restrict__ out_lab) {
  const int t = blockIdx.x * 256 + threadIdx.x;
  if (t >= BB * NPOINTS) return;
  const int b = t / NPOINTS, q = t % NPOINTS;
  const int id = fpsIdx[t];
  #pragma unroll
  for (int c = 0; c < 3; ++c) {
    float v = xyz[((size_t)b * 3 + c) * NN + id];
    newxyz[(size_t)t * 3 + c] = v;
    out_xyz[((size_t)b * 3 + c) * NPOINTS + q] = v;
  }
  out_lab[t] = (float)labels[(size_t)b * NN + id];
}

// ---------------------------------------------------------------------------
// KNN top-32 by iterative min extraction
// ---------------------------------------------------------------------------
__global__ void __launch_bounds__(256) knn_kernel(const float* __restrict__ xyz,
                                                  const float* __restrict__ newxyz,
                                                  int* __restrict__ knnIdx) {
  const int bq = blockIdx.x;
  const int b = bq / NPOINTS;
  const int tid = threadIdx.x;
  __shared__ float d[NN];
  __shared__ float pv[8];
  __shared__ int   pi[8];
  const float qx = newxyz[(size_t)bq * 3 + 0];
  const float qy = newxyz[(size_t)bq * 3 + 1];
  const float qz = newxyz[(size_t)bq * 3 + 2];
  const float* xb = xyz + (size_t)b * 3 * NN;
  for (int i = tid; i < NN; i += 256) {
    float dx = xb[i] - qx, dy = xb[NN + i] - qy, dz = xb[2 * NN + i] - qz;
    d[i] = dx * dx + dy * dy + dz * dz;
  }
  __syncthreads();
  const int wv = tid >> 5, ln = tid & 31;
  for (int s = 0; s < NSAMP; ++s) {
    float bv = 3.0e38f; int bi = 0;
    for (int i = tid; i < NN; i += 256) {
      float v = d[i];
      if (v < bv) { bv = v; bi = i; }
    }
    #pragma unroll
    for (int off = 16; off > 0; off >>= 1) {
      float ov = __shfl_xor(bv, off, 32);
      int   oi = __shfl_xor(bi, off, 32);
      if (ov < bv || (ov == bv && oi < bi)) { bv = ov; bi = oi; }
    }
    if (ln == 0) { pv[wv] = bv; pi[wv] = bi; }
    __syncthreads();
    if (tid == 0) {
      float v2 = pv[0]; int i2 = pi[0];
      for (int i = 1; i < 8; ++i)
        if (pv[i] < v2 || (pv[i] == v2 && pi[i] < i2)) { v2 = pv[i]; i2 = pi[i]; }
      knnIdx[(size_t)bq * NSAMP + s] = i2;
      d[i2] = 3.0e38f;
    }
    __syncthreads();
  }
}

// ---------------------------------------------------------------------------
// Grouping: x0 (B,96pad,32,1024) and w0 (B,32pad,32,1024)
// ---------------------------------------------------------------------------
__global__ void group_kernel(const float* __restrict__ xyz, const float* __restrict__ points,
                             const float* __restrict__ dscale, const float* __restrict__ newxyz,
                             const int* __restrict__ knnIdx, float* __restrict__ x0,
                             float* __restrict__ w0) {
  const int t = blockIdx.x * 256 + threadIdx.x;
  if (t >= BB * NPOINTS * NSAMP) return;
  const int b = t / (NPOINTS * NSAMP);
  const int r = t % (NPOINTS * NSAMP);
  const int s = r / NSAMP, k = r % NSAMP;
  const int id = knnIdx[t];
  const int bq = b * NPOINTS + s;
  const float gx = xyz[((size_t)b * 3 + 0) * NN + id] - newxyz[(size_t)bq * 3 + 0];
  const float gy = xyz[((size_t)b * 3 + 1) * NN + id] - newxyz[(size_t)bq * 3 + 1];
  const float gz = xyz[((size_t)b * 3 + 2) * NN + id] - newxyz[(size_t)bq * 3 + 2];
  const int pos = k * NPOINTS + s;
  float* xb = x0 + (size_t)b * CIN0P * KS;
  xb[0 * KS + pos] = gx;
  xb[1 * KS + pos] = gy;
  xb[2 * KS + pos] = gz;
  const float* pb = points + (size_t)b * DD * NN;
  for (int c = 0; c < DD; ++c) xb[(size_t)(3 + c) * KS + pos] = pb[(size_t)c * NN + id];
  const float dsv = dscale[(size_t)b * NN + id];
  float* wb = w0 + (size_t)b * 32 * KS;
  wb[0 * KS + pos] = gx * dsv;
  wb[1 * KS + pos] = gy * dsv;
  wb[2 * KS + pos] = gz * dsv;
}

// ---------------------------------------------------------------------------
// WMMA GEMM: out[b][o][n] = bias[o] + sum_c Wh[o][c] * actT[b][n][c]
// Wh: f16 (Opad x Cpad), actT: f16 (B, Npb, Cpad) -- both zero padded.
// One wave -> 16x64 strip; per k-step: 2 b128 A loads + 8 b128 B loads + 4 WMMA.
// ---------------------------------------------------------------------------
__global__ void __launch_bounds__(128) wmma_gemm_kernel(
    const _Float16* __restrict__ Wh, const float* __restrict__ bias,
    const _Float16* __restrict__ actT, float* __restrict__ out,
    int O, int Ostride, int Cpad, int Npb) {
  const int wid  = blockIdx.x * (blockDim.x >> 5) + (threadIdx.x >> 5);
  const int lane = threadIdx.x & 31;
  const int tilesM  = (O + 15) >> 4;
  const int tilesN4 = Npb >> 6;
  const int total   = BB * tilesM * tilesN4;
  if (wid >= total) return;                       // uniform per wave
  const int b   = wid / (tilesM * tilesN4);
  const int rr  = wid % (tilesM * tilesN4);
  const int mt  = rr / tilesN4, nt4 = rr % tilesN4;
  const int l16 = lane & 15;
  const int hi  = lane >> 4;
  const int arow  = mt * 16 + l16;
  const int akoff = hi ? 16 : 0;
  const int bkoff = hi ? 16 : 0;
  const int col0  = nt4 * 64 + l16;
  const _Float16* wrow = Wh + (size_t)arow * Cpad + akoff;
  const _Float16* actb = actT + (size_t)b * Npb * Cpad + bkoff;
  const _Float16* c0p = actb + (size_t)(col0 +  0) * Cpad;
  const _Float16* c1p = actb + (size_t)(col0 + 16) * Cpad;
  const _Float16* c2p = actb + (size_t)(col0 + 32) * Cpad;
  const _Float16* c3p = actb + (size_t)(col0 + 48) * Cpad;
  v8f acc0 = {}, acc1 = {}, acc2 = {}, acc3 = {};
  for (int kb = 0; kb < Cpad; kb += 32) {
    const v8h alo = *(const v8h*)(wrow + kb);
    const v8h ahi = *(const v8h*)(wrow + kb + 8);
    const v16h a = __builtin_shufflevector(alo, ahi, 0, 1, 2, 3, 4, 5, 6, 7,
                                           8, 9, 10, 11, 12, 13, 14, 15);
    {
      const v8h blo = *(const v8h*)(c0p + kb);
      const v8h bhi = *(const v8h*)(c0p + kb + 8);
      const v16h bm = __builtin_shufflevector(blo, bhi, 0, 1, 2, 3, 4, 5, 6, 7,
                                              8, 9, 10, 11, 12, 13, 14, 15);
      acc0 = __builtin_amdgcn_wmma_f32_16x16x32_f16(false, a, false, bm, (short)0, acc0, false, false);
    }
    {
      const v8h blo = *(const v8h*)(c1p + kb);
      const v8h bhi = *(const v8h*)(c1p + kb + 8);
      const v16h bm = __builtin_shufflevector(blo, bhi, 0, 1, 2, 3, 4, 5, 6, 7,
                                              8, 9, 10, 11, 12, 13, 14, 15);
      acc1 = __builtin_amdgcn_wmma_f32_16x16x32_f16(false, a, false, bm, (short)0, acc1, false, false);
    }
    {
      const v8h blo = *(const v8h*)(c2p + kb);
      const v8h bhi = *(const v8h*)(c2p + kb + 8);
      const v16h bm = __builtin_shufflevector(blo, bhi, 0, 1, 2, 3, 4, 5, 6, 7,
                                              8, 9, 10, 11, 12, 13, 14, 15);
      acc2 = __builtin_amdgcn_wmma_f32_16x16x32_f16(false, a, false, bm, (short)0, acc2, false, false);
    }
    {
      const v8h blo = *(const v8h*)(c3p + kb);
      const v8h bhi = *(const v8h*)(c3p + kb + 8);
      const v16h bm = __builtin_shufflevector(blo, bhi, 0, 1, 2, 3, 4, 5, 6, 7,
                                              8, 9, 10, 11, 12, 13, 14, 15);
      acc3 = __builtin_amdgcn_wmma_f32_16x16x32_f16(false, a, false, bm, (short)0, acc3, false, false);
    }
  }
  float* outb = out + (size_t)b * Ostride * Npb;
  const int rbase = mt * 16 + hi * 8;
  #pragma unroll
  for (int r8 = 0; r8 < 8; ++r8) {
    const int row = rbase + r8;
    if (row < O) {
      const float bv = bias[row];
      float* orow = outb + (size_t)row * Npb + col0;
      orow[0]  = acc0[r8] + bv;
      orow[16] = acc1[r8] + bv;
      orow[32] = acc2[r8] + bv;
      orow[48] = acc3[r8] + bv;
    }
  }
}

// ---------------------------------------------------------------------------
// feat_t[b][c*16+w][s] = sum_k x3[b][c][k][s] * w3[b][w][k][s]
// ---------------------------------------------------------------------------
__global__ void __launch_bounds__(128) feat_einsum_kernel(
    const float* __restrict__ x3, const float* __restrict__ w3,
    float* __restrict__ feat_t) {
  const int wid  = blockIdx.x * (blockDim.x >> 5) + (threadIdx.x >> 5);
  const int lane = threadIdx.x & 31;
  const int total = BB * NPOINTS * 8;
  if (wid >= total) return;
  const int b  = wid / (NPOINTS * 8);
  const int rr = wid % (NPOINTS * 8);
  const int s  = rr / 8, mt = rr % 8;
  const int l16 = lane & 15;
  const int hi  = lane >> 4;
  const int arow  = mt * 16 + l16;
  const int akoff = hi ? 16 : 0;
  const int bkoff = hi ? 16 : 0;
  const float* xb = x3 + (size_t)b * 128 * KS + (size_t)arow * KS + s;
  const float* wb = w3 + (size_t)b * 16 * KS + (size_t)l16 * KS + s;
  v16h a, bm;
  #pragma unroll
  for (int j = 0; j < 4; ++j) {
    int k0 = akoff + 2 * j;
    int k1 = akoff + 8 + 2 * j;
    a[2 * j]         = (_Float16)xb[(size_t)k0 * NPOINTS];
    a[2 * j + 1]     = (_Float16)xb[(size_t)(k0 + 1) * NPOINTS];
    a[8 + 2 * j]     = (_Float16)xb[(size_t)k1 * NPOINTS];
    a[8 + 2 * j + 1] = (_Float16)xb[(size_t)(k1 + 1) * NPOINTS];
  }
  #pragma unroll
  for (int j = 0; j < 8; ++j) {
    int k0 = bkoff + 2 * j;
    bm[2 * j]     = (_Float16)wb[(size_t)k0 * NPOINTS];
    bm[2 * j + 1] = (_Float16)wb[(size_t)(k0 + 1) * NPOINTS];
  }
  v8f acc = {};
  acc = __builtin_amdgcn_wmma_f32_16x16x32_f16(false, a, false, bm, (short)0, acc,
                                               false, false);
  float* fb = feat_t + (size_t)b * 2048 * NPOINTS;
  const int rbase = mt * 16 + hi * 8;
  #pragma unroll
  for (int r8 = 0; r8 < 8; ++r8) {
    const int rowc = rbase + r8;
    fb[(size_t)(rowc * 16 + l16) * NPOINTS + s] = acc[r8];
  }
}

// ---------------------------------------------------------------------------
// BatchNorm stats / apply (+ReLU); Cs = channel stride of padded layout
// ---------------------------------------------------------------------------
__global__ void __launch_bounds__(256) bn_stats_kernel(const float* __restrict__ buf,
                                                       float* __restrict__ stats,
                                                       int Cs, int Npb) {
  const int o = blockIdx.x;
  const int tid = threadIdx.x;
  const int count = BB * Npb;
  float s = 0.0f, s2 = 0.0f;
  for (int i = tid; i < count; i += 256) {
    const int b = i / Npb, n = i - b * Npb;
    float v = buf[((size_t)b * Cs + o) * Npb + n];
    s += v; s2 += v * v;
  }
  #pragma unroll
  for (int off = 16; off > 0; off >>= 1) {
    s  += __shfl_xor(s, off, 32);
    s2 += __shfl_xor(s2, off, 32);
  }
  __shared__ float ps[8], ps2[8];
  const int wv = tid >> 5, ln = tid & 31;
  if (ln == 0) { ps[wv] = s; ps2[wv] = s2; }
  __syncthreads();
  if (tid == 0) {
    float S = 0.0f, S2 = 0.0f;
    for (int i = 0; i < 8; ++i) { S += ps[i]; S2 += ps2[i]; }
    float m = S / (float)count;
    stats[2 * o] = m;
    stats[2 * o + 1] = S2 / (float)count - m * m;
  }
}

__global__ void bn_apply_kernel(const float* __restrict__ in, float* __restrict__ out,
                                const float* __restrict__ stats, const float* __restrict__ g,
                                const float* __restrict__ be, int C, int CsIn, int CsOut,
                                int Npb, size_t total) {
  const size_t idx = (size_t)blockIdx.x * 256 + threadIdx.x;
  if (idx >= total) return;
  const int n = (int)(idx % (size_t)Npb);
  const int o = (int)((idx / (size_t)Npb) % (size_t)C);
  const int b = (int)(idx / ((size_t)Npb * C));
  const float m = stats[2 * o], v = stats[2 * o + 1];
  const float x = (in[((size_t)b * CsIn + o) * Npb + n] - m) * rsqrtf(v + EPSV) * g[o] + be[o];
  out[((size_t)b * CsOut + o) * Npb + n] = fmaxf(x, 0.0f);
}

// ---------------------------------------------------------------------------
extern "C" void kernel_launch(void* const* d_in, const int* in_sizes, int n_in,
                              void* d_out, int out_size, void* d_ws, size_t ws_size,
                              hipStream_t stream) {
  (void)in_sizes; (void)n_in; (void)out_size; (void)ws_size;
  const float* xyz    = (const float*)d_in[0];
  const float* points = (const float*)d_in[1];
  const int*   labels = (const int*)d_in[2];
  int p = 3;
  const float *dW[3], *db[3], *dg[3], *dbe[3];
  const float *mW[3], *mb[3], *mg[3], *mbe[3];
  const float *wW[3], *wb_[3], *wg[3], *wbe[3];
  for (int i = 0; i < 3; ++i) { dW[i] = (const float*)d_in[p++]; db[i] = (const float*)d_in[p++];
                                dg[i] = (const float*)d_in[p++]; dbe[i] = (const float*)d_in[p++]; }
  for (int i = 0; i < 3; ++i) { mW[i] = (const float*)d_in[p++]; mb[i] = (const float*)d_in[p++];
                                mg[i] = (const float*)d_in[p++]; mbe[i] = (const float*)d_in[p++]; }
  for (int i = 0; i < 3; ++i) { wW[i] = (const float*)d_in[p++]; wb_[i] = (const float*)d_in[p++];
                                wg[i] = (const float*)d_in[p++]; wbe[i] = (const float*)d_in[p++]; }
  const float* linW = (const float*)d_in[p++];
  const float* linb = (const float*)d_in[p++];
  const float* blg  = (const float*)d_in[p++];
  const float* blb  = (const float*)d_in[p++];

  float* ws = (float*)d_ws;
  size_t off = 0;
  auto alloc = [&](size_t n) { size_t r = off; off += (n + 63) & ~(size_t)63; return r; };
  float*     f_dens   = ws + alloc((size_t)BB * 32 * NN);
  float*     f_dsA    = ws + alloc((size_t)BB * 32 * NN);
  float*     f_dsB    = ws + alloc((size_t)BB * 32 * NN);
  float*     f_dscale = ws + alloc((size_t)BB * NN);
  int*       i_fps    = (int*)(ws + alloc((size_t)BB * NPOINTS));
  float*     f_newxyz = ws + alloc((size_t)BB * NPOINTS * 3);
  int*       i_knn    = (int*)(ws + alloc((size_t)BB * NPOINTS * NSAMP));
  float*     bufA     = ws + alloc((size_t)BB * CIN0P * KS);  // x0(96p), x2(64), feat_t
  float*     bufB     = ws + alloc((size_t)BB * 128 * KS);    // x1(64), x3(128)
  float*     bufC     = ws + alloc((size_t)BB * 32 * KS);     // w0(32p), w2(32p)
  float*     bufD     = ws + alloc((size_t)BB * 32 * KS);     // w1(32p)
  float*     f_w3     = ws + alloc((size_t)BB * 16 * KS);     // w3 exact 16ch
  float*     f_lin    = ws + alloc((size_t)BB * 128 * NPOINTS);
  float*     f_stats  = ws + alloc(512);
  _Float16*  f_wh     = (_Float16*)(ws + alloc(140000));            // staged weights f16
  _Float16*  f_act16  = (_Float16*)(ws + alloc((size_t)BB * CIN0P * KS / 2 + 64)); // staged acts f16

  float* out      = (float*)d_out;
  float* out_xyz  = out;
  float* out_feat = out + (size_t)BB * 3 * NPOINTS;
  float* out_lab  = out + (size_t)BB * 3 * NPOINTS + (size_t)BB * 128 * NPOINTS;

  auto zero = [&](float* ptr, size_t n) {
    zero_kernel<<<(unsigned)((n + 255) / 256), 256, 0, stream>>>(ptr, n);
  };
  auto gemm = [&](const float* Wt, const float* bias, const float* inF, float* o_,
                  int O, int Ostride, int C, int Cpad, int Npb) {
    int Opad = ((O + 15) / 16) * 16;
    prep_weights_kernel<<<(Opad * Cpad + 255) / 256, 256, 0, stream>>>(Wt, f_wh, O, C, Opad, Cpad);
    act_transpose_kernel<<<dim3(Npb / 32, Cpad / 32, BB), dim3(32, 8), 0, stream>>>(
        inF, f_act16, Cpad, Npb);
    int waves = BB * (Opad / 16) * (Npb / 64);
    wmma_gemm_kernel<<<(waves + 3) / 4, 128, 0, stream>>>(f_wh, bias, f_act16, o_,
                                                          O, Ostride, Cpad, Npb);
  };
  auto bn = [&](const float* in, float* o_, const float* g, const float* be,
                int C, int Cs, int CsOut, int Npb) {
    bn_stats_kernel<<<C, 256, 0, stream>>>(in, f_stats, Cs, Npb);
    size_t total = (size_t)BB * C * Npb;
    bn_apply_kernel<<<(unsigned)((total + 255) / 256), 256, 0, stream>>>(
        in, o_, f_stats, g, be, C, Cs, CsOut, Npb, total);
  };

  // 0) zero padded buffers
  zero(f_dens, (size_t)BB * 32 * NN);
  zero(f_dsA,  (size_t)BB * 32 * NN);
  zero(f_dsB,  (size_t)BB * 32 * NN);
  zero(bufA,   (size_t)BB * CIN0P * KS);
  zero(bufC,   (size_t)BB * 32 * KS);
  zero(bufD,   (size_t)BB * 32 * KS);

  // 1) kernel density
  density_kernel<<<dim3(NN / 256, BB), 256, 0, stream>>>(xyz, f_dens);
  // 2) density MLP 1 -> 8 -> 8 -> 1
  gemm(dW[0], db[0], f_dens, f_dsA, 8, 32, 1, 32, NN);
  bn(f_dsA, f_dsA, dg[0], dbe[0], 8, 32, 32, NN);
  gemm(dW[1], db[1], f_dsA, f_dsB, 8, 32, 8, 32, NN);
  bn(f_dsB, f_dsB, dg[1], dbe[1], 8, 32, 32, NN);
  gemm(dW[2], db[2], f_dsB, f_dscale, 1, 1, 8, 32, NN);
  bn(f_dscale, f_dscale, dg[2], dbe[2], 1, 1, 1, NN);
  // 3) FPS + gathers
  fps_kernel<<<BB, 1024, 0, stream>>>(xyz, i_fps);
  gather_fps_kernel<<<(BB * NPOINTS + 255) / 256, 256, 0, stream>>>(
      xyz, labels, i_fps, f_newxyz, out_xyz, out_lab);
  // 4) KNN
  knn_kernel<<<BB * NPOINTS, 256, 0, stream>>>(xyz, f_newxyz, i_knn);
  // 5) grouping
  group_kernel<<<(BB * NPOINTS * NSAMP + 255) / 256, 256, 0, stream>>>(
      xyz, points, f_dscale, f_newxyz, i_knn, bufA, bufC);
  // 6) point MLP 67 -> 64 -> 64 -> 128
  gemm(mW[0], mb[0], bufA, bufB, 64, 64, CIN0, CIN0P, KS);
  bn(bufB, bufB, mg[0], mbe[0], 64, 64, 64, KS);
  gemm(mW[1], mb[1], bufB, bufA, 64, 64, 64, 64, KS);
  bn(bufA, bufA, mg[1], mbe[1], 64, 64, 64, KS);
  gemm(mW[2], mb[2], bufA, bufB, 128, 128, 64, 64, KS);
  bn(bufB, bufB, mg[2], mbe[2], 128, 128, 128, KS);
  // 7) weight net 3 -> 8 -> 8 -> 16
  gemm(wW[0], wb_[0], bufC, bufD, 8, 32, 3, 32, KS);
  bn(bufD, bufD, wg[0], wbe[0], 8, 32, 32, KS);
  gemm(wW[1], wb_[1], bufD, bufC, 8, 32, 8, 32, KS);
  bn(bufC, bufC, wg[1], wbe[1], 8, 32, 32, KS);
  gemm(wW[2], wb_[2], bufC, f_w3, 16, 16, 8, 32, KS);
  bn(f_w3, f_w3, wg[2], wbe[2], 16, 16, 16, KS);
  // 8) per-point einsum -> feat_t in bufA
  {
    int waves = BB * NPOINTS * 8;
    feat_einsum_kernel<<<(waves + 3) / 4, 128, 0, stream>>>(bufB, f_w3, bufA);
  }
  // 9) linear 2048 -> 128, bn_lin + ReLU -> d_out
  gemm(linW, linb, bufA, f_lin, 128, 128, 2048, 2048, NPOINTS);
  bn(f_lin, out_feat, blg, blb, 128, 128, 128, NPOINTS);
}
